// Superpoint_MAE_29265907155076
// MI455X (gfx1250) — matrix operand order repaired and checked by reading, not compiled, not run on hardware
//
#include <hip/hip_runtime.h>
#include <hip/hip_bf16.h>

typedef __attribute__((ext_vector_type(16))) __bf16 v16bf;
typedef __attribute__((ext_vector_type(8)))  __bf16 v8bf;
typedef __attribute__((ext_vector_type(8)))  float  v8f;

#define N_SEG  4096
#define C_IN   11
#define C_H1   128
#define C_H2   256
#define C_CAT  512
#define C_Z    512
#define C_OUT  384

#define TA 128           // points per block, pass A
#define TB 64            // points per block, pass B
#define H1S (C_H1 + 8)   // padded LDS strides (bank-conflict-free A-frag loads)
#define HCS (C_CAT + 8)
#define ZS  (C_Z + 8)

// ---- workspace layout (bytes) ----
#define G_BYTES   ((size_t)N_SEG * C_H2 * 4)      // 4 MiB: keys during pass A, f32 after decode
#define W2T_OFF   (G_BYTES)
#define W2T_BYTES ((size_t)C_H2 * C_H1 * 2)       // bf16, [N][K] column-major
#define W3T_OFF   (W2T_OFF + W2T_BYTES)
#define W3T_BYTES ((size_t)C_Z * C_CAT * 2)
#define W4T_OFF   (W3T_OFF + W3T_BYTES)
#define W4T_BYTES ((size_t)C_OUT * C_Z * 2)

// ---- order-preserving f32 <-> u32 key (monotonic; single atomic_max_u32, no branches) ----
#define KEY_NEG_INF 0x007FFFFFu                   // enc(-inf)
__device__ __forceinline__ unsigned enc_f32(float f) {
    unsigned u = __float_as_uint(f);
    return (u & 0x80000000u) ? ~u : (u | 0x80000000u);
}
__device__ __forceinline__ float dec_key(unsigned k) {
    return __uint_as_float((k & 0x80000000u) ? (k & 0x7FFFFFFFu) : ~k);
}

// ---------------- fragment helpers (CDNA5 16-bit WMMA layouts) ----------------
// A 16x32 bf16: lane<16 -> row=lane, K = e<8 ? e : 16+(e-8); lane>=16 -> row=lane-16, K += 8
__device__ __forceinline__ v16bf load_a_frag_lds(const __bf16* base, int stride, int m0, int k0) {
    int lane = threadIdx.x & 31;
    int half = lane >> 4;
    const __bf16* p = base + (size_t)(m0 + (lane & 15)) * stride + k0 + half * 8;
    v8bf lo = *(const v8bf*)p;          // K = half*8 + 0..7
    v8bf hi = *(const v8bf*)(p + 16);   // K = 16 + half*8 + 0..7
    v16bf a;
#pragma unroll
    for (int e = 0; e < 8; ++e) { a[e] = lo[e]; a[e + 8] = hi[e]; }
    return a;
}

// B 32x16 bf16 from global weights stored column-major [N][K]: lane -> col, K = e<8 ? half*8+e : 16+half*8+(e-8)
__device__ __forceinline__ v16bf load_b_frag_g(const __bf16* wt, int ldk, int n0, int k0) {
    int lane = threadIdx.x & 31;
    int half = lane >> 4;
    const __bf16* p = wt + (size_t)(n0 + (lane & 15)) * ldk + k0 + half * 8;
    v8bf lo = *(const v8bf*)p;
    v8bf hi = *(const v8bf*)(p + 16);
    v16bf b;
#pragma unroll
    for (int e = 0; e < 8; ++e) { b[e] = lo[e]; b[e + 8] = hi[e]; }
    return b;
}

// ---------------- init keys to enc(-inf) ----------------
__global__ __launch_bounds__(256) void k_init(unsigned* __restrict__ gk, unsigned* __restrict__ outk) {
    int i = blockIdx.x * 256 + threadIdx.x;
    if (i < N_SEG * C_H2)  gk[i]   = KEY_NEG_INF;
    if (i < N_SEG * C_OUT) outk[i] = KEY_NEG_INF;
}

// ---------------- decode keys -> floats, in place ----------------
__global__ __launch_bounds__(256) void k_decode(unsigned* __restrict__ buf, int count) {
    int i = blockIdx.x * 256 + threadIdx.x;
    if (i < count) {
        unsigned k = buf[i];
        ((float*)buf)[i] = dec_key(k);
    }
}

// ---------------- weights: f32 row-major -> bf16 column-major [N][K] ----------------
__global__ __launch_bounds__(256) void k_prep(const float* __restrict__ W2,
                                              const float* __restrict__ W3,
                                              const float* __restrict__ W4,
                                              __bf16* __restrict__ W2t,
                                              __bf16* __restrict__ W3t,
                                              __bf16* __restrict__ W4t) {
    int i = blockIdx.x * 256 + threadIdx.x;
    if (i < C_H2 * C_H1) { int nn = i / C_H1, k = i % C_H1; W2t[i] = (__bf16)W2[k * C_H2 + nn]; }
    if (i < C_Z * C_CAT) { int nn = i / C_CAT, k = i % C_CAT; W3t[i] = (__bf16)W3[k * C_Z + nn]; }
    if (i < C_OUT * C_Z) { int nn = i / C_Z,  k = i % C_Z;  W4t[i] = (__bf16)W4[k * C_OUT + nn]; }
}

// ---------------- Pass A: h = (X@W1+b1)@W2+b2 ; atomic segment-max keys into gk ----------------
// Wave split: each wave owns 2 n-tiles and ALL 8 m-tiles -> every B fragment loaded once per block.
__global__ __launch_bounds__(256) void k_passA(const float* __restrict__ feat,
                                               const int* __restrict__ sidx, int n,
                                               const float* __restrict__ W1,
                                               const float* __restrict__ b1,
                                               const __bf16* __restrict__ W2t,
                                               const float* __restrict__ b2,
                                               unsigned* __restrict__ gk) {
    __shared__ float  s_feat[TA][12];
    __shared__ float  s_W1[C_IN * C_H1];
    __shared__ float  s_b1[C_H1];
    __shared__ int    s_seg[TA];
    __shared__ __bf16 s_h1[TA * H1S];

    const int t = threadIdx.x;
    const int base = blockIdx.x * TA;

    for (int i = t; i < C_IN * C_H1; i += 256) s_W1[i] = W1[i];
    if (t < C_H1) s_b1[t] = b1[t];
    for (int i = t; i < TA * C_IN; i += 256) {
        int m = i / C_IN, k = i % C_IN, gm = base + m;
        s_feat[m][k] = (gm < n) ? feat[(size_t)gm * C_IN + k] : 0.0f;
    }
    if (t < TA) { int gm = base + t; s_seg[t] = (gm < n) ? sidx[gm] : 0; }
    __syncthreads();

    // GEMM1 (K=11) scalar f32 -> bf16 h1 tile
#pragma unroll 4
    for (int i = 0; i < (TA * C_H1) / 256; ++i) {
        int lin = t + i * 256;
        int m = lin >> 7, c = lin & 127;
        float acc = s_b1[c];
#pragma unroll
        for (int k = 0; k < C_IN; ++k) acc += s_feat[m][k] * s_W1[k * C_H1 + c];
        s_h1[m * H1S + c] = (__bf16)acc;
    }
    __syncthreads();

    const int wave = t >> 5, lane = t & 31, half = lane >> 4;

    v8f acc[16];                 // [mt][j] -> acc[mt*2+j]
#pragma unroll
    for (int i = 0; i < 16; ++i) acc[i] = (v8f){};

    for (int k = 0; k < 4; ++k) {
        v16bf af[8];
#pragma unroll
        for (int mt = 0; mt < 8; ++mt) af[mt] = load_a_frag_lds(s_h1, H1S, mt * 16, k * 32);
#pragma unroll
        for (int j = 0; j < 2; ++j) {
            v16bf bfr = load_b_frag_g(W2t, C_H1, (wave * 2 + j) * 16, k * 32);
#pragma unroll
            for (int mt = 0; mt < 8; ++mt)
                acc[mt * 2 + j] = __builtin_amdgcn_wmma_f32_16x16x32_bf16(
                    false, af[mt], false, bfr, (short)0, acc[mt * 2 + j], false, false);
        }
    }

#pragma unroll
    for (int j = 0; j < 2; ++j) {
        const int nn = (wave * 2 + j) * 16 + (lane & 15);
        const float bias = b2[nn];
#pragma unroll
        for (int mt = 0; mt < 8; ++mt) {
#pragma unroll
            for (int r = 0; r < 8; ++r) {
                int m = mt * 16 + r + 8 * half;
                if (base + m < n)
                    atomicMax(&gk[(unsigned)s_seg[m] * C_H2 + nn],
                              enc_f32(acc[mt * 2 + j][r] + bias));
            }
        }
    }
}

// ---- Pass B LDS carve offsets (bytes) ----
#define SM_SEG   0
#define SM_FEAT  256
#define SM_W1    3328
#define SM_B1    8960
#define SM_H1    9472
#define SM_HCAT  26880
#define SM_Z     93440
#define SM_TOTAL 160000

// -------- Pass B: recompute h, gather g, z=relu(hcat@W3+b3)@W4+b4, seg-max keys into outk --------
// Wave split everywhere: disjoint N slice per wave, all 4 m-tiles -> B fragments loaded once/block.
__global__ __launch_bounds__(256) void k_passB(const float* __restrict__ feat,
                                               const int* __restrict__ sidx, int n,
                                               const float* __restrict__ W1,
                                               const float* __restrict__ b1,
                                               const __bf16* __restrict__ W2t,
                                               const float* __restrict__ b2,
                                               const float* __restrict__ g,
                                               const __bf16* __restrict__ W3t,
                                               const float* __restrict__ b3,
                                               const __bf16* __restrict__ W4t,
                                               const float* __restrict__ b4,
                                               unsigned* __restrict__ outk) {
    extern __shared__ char smem[];
    int*    s_seg  = (int*)(smem + SM_SEG);
    float*  s_feat = (float*)(smem + SM_FEAT);   // [TB][12]
    float*  s_W1   = (float*)(smem + SM_W1);
    float*  s_b1   = (float*)(smem + SM_B1);
    __bf16* s_h1   = (__bf16*)(smem + SM_H1);    // [TB][H1S]
    __bf16* s_hcat = (__bf16*)(smem + SM_HCAT);  // [TB][HCS]: cols 0..255 = g, 256..511 = h
    __bf16* s_z    = (__bf16*)(smem + SM_Z);     // [TB][ZS]

    const int t = threadIdx.x;
    const int base = blockIdx.x * TB;

    for (int i = t; i < C_IN * C_H1; i += 256) s_W1[i] = W1[i];
    if (t < C_H1) s_b1[t] = b1[t];
    for (int i = t; i < TB * C_IN; i += 256) {
        int m = i / C_IN, k = i % C_IN, gm = base + m;
        s_feat[m * 12 + k] = (gm < n) ? feat[(size_t)gm * C_IN + k] : 0.0f;
    }
    if (t < TB) { int gm = base + t; s_seg[t] = (gm < n) ? sidx[gm] : 0; }
    __syncthreads();

    // GEMM1 recompute
#pragma unroll 4
    for (int i = 0; i < (TB * C_H1) / 256; ++i) {
        int lin = t + i * 256;
        int m = lin >> 7, c = lin & 127;
        float acc = s_b1[c];
#pragma unroll
        for (int k = 0; k < C_IN; ++k) acc += s_feat[m * 12 + k] * s_W1[k * C_H1 + c];
        s_h1[m * H1S + c] = (__bf16)acc;
    }
    __syncthreads();

    const int wave = t >> 5, lane = t & 31, half = lane >> 4;

    // ---- GEMM2 -> hcat cols [256..511]; wave: n-tiles {2w,2w+1}, all 4 m-tiles ----
    {
        v8f acc2[8];
#pragma unroll
        for (int i = 0; i < 8; ++i) acc2[i] = (v8f){};
        for (int k = 0; k < 4; ++k) {
            v16bf af[4];
#pragma unroll
            for (int mt = 0; mt < 4; ++mt) af[mt] = load_a_frag_lds(s_h1, H1S, mt * 16, k * 32);
#pragma unroll
            for (int j = 0; j < 2; ++j) {
                v16bf bfr = load_b_frag_g(W2t, C_H1, (wave * 2 + j) * 16, k * 32);
#pragma unroll
                for (int mt = 0; mt < 4; ++mt)
                    acc2[mt * 2 + j] = __builtin_amdgcn_wmma_f32_16x16x32_bf16(
                        false, af[mt], false, bfr, (short)0, acc2[mt * 2 + j], false, false);
            }
        }
#pragma unroll
        for (int j = 0; j < 2; ++j) {
            const int nn = (wave * 2 + j) * 16 + (lane & 15);
            const float bias = b2[nn];
#pragma unroll
            for (int mt = 0; mt < 4; ++mt)
#pragma unroll
                for (int r = 0; r < 8; ++r) {
                    int m = mt * 16 + r + 8 * half;
                    s_hcat[m * HCS + 256 + nn] = (__bf16)(acc2[mt * 2 + j][r] + bias);
                }
        }
    }
    // gather g rows -> hcat cols [0..255] (L2-resident, 64 f32 per thread)
    {
        const int row = t >> 2, q = t & 3;
        const float* grow = g + (size_t)s_seg[row] * C_H2 + q * 64;
        __bf16* dst = s_hcat + row * HCS + q * 64;
#pragma unroll
        for (int j = 0; j < 64; j += 4) {
            float4 v = *(const float4*)(grow + j);
            dst[j + 0] = (__bf16)v.x; dst[j + 1] = (__bf16)v.y;
            dst[j + 2] = (__bf16)v.z; dst[j + 3] = (__bf16)v.w;
        }
    }
    __syncthreads();

    // ---- GEMM3 + ReLU -> s_z; wave: n-tiles {4w..4w+3}, all 4 m-tiles ----
    {
        v8f acc3[16];
#pragma unroll
        for (int i = 0; i < 16; ++i) acc3[i] = (v8f){};
        for (int k = 0; k < 16; ++k) {
            v16bf af[4];
#pragma unroll
            for (int mt = 0; mt < 4; ++mt) af[mt] = load_a_frag_lds(s_hcat, HCS, mt * 16, k * 32);
#pragma unroll
            for (int j = 0; j < 4; ++j) {
                v16bf bfr = load_b_frag_g(W3t, C_CAT, (wave * 4 + j) * 16, k * 32);
#pragma unroll
                for (int mt = 0; mt < 4; ++mt)
                    acc3[mt * 4 + j] = __builtin_amdgcn_wmma_f32_16x16x32_bf16(
                        false, af[mt], false, bfr, (short)0, acc3[mt * 4 + j], false, false);
            }
        }
#pragma unroll
        for (int j = 0; j < 4; ++j) {
            const int nn = (wave * 4 + j) * 16 + (lane & 15);
            const float bias = b3[nn];
#pragma unroll
            for (int mt = 0; mt < 4; ++mt)
#pragma unroll
                for (int r = 0; r < 8; ++r) {
                    int m = mt * 16 + r + 8 * half;
                    float v = acc3[mt * 4 + j][r] + bias;
                    s_z[m * ZS + nn] = (__bf16)(v > 0.0f ? v : 0.0f);
                }
        }
    }
    __syncthreads();

    // ---- GEMM4 -> atomic segment-max keys; wave: n-tiles {3w..3w+2}, all 4 m-tiles ----
    {
        v8f acc4[12];
#pragma unroll
        for (int i = 0; i < 12; ++i) acc4[i] = (v8f){};
        for (int k = 0; k < 16; ++k) {
            v16bf af[4];
#pragma unroll
            for (int mt = 0; mt < 4; ++mt) af[mt] = load_a_frag_lds(s_z, ZS, mt * 16, k * 32);
#pragma unroll
            for (int j = 0; j < 3; ++j) {
                v16bf bfr = load_b_frag_g(W4t, C_Z, (wave * 3 + j) * 16, k * 32);
#pragma unroll
                for (int mt = 0; mt < 4; ++mt)
                    acc4[mt * 3 + j] = __builtin_amdgcn_wmma_f32_16x16x32_bf16(
                        false, af[mt], false, bfr, (short)0, acc4[mt * 3 + j], false, false);
            }
        }
#pragma unroll
        for (int j = 0; j < 3; ++j) {
            const int nn = (wave * 3 + j) * 16 + (lane & 15);
            const float bias = b4[nn];
#pragma unroll
            for (int mt = 0; mt < 4; ++mt)
#pragma unroll
                for (int r = 0; r < 8; ++r) {
                    int m = mt * 16 + r + 8 * half;
                    if (base + m < n)
                        atomicMax(&outk[(unsigned)s_seg[m] * C_OUT + nn],
                                  enc_f32(acc4[mt * 3 + j][r] + bias));
                }
        }
    }
}

extern "C" void kernel_launch(void* const* d_in, const int* in_sizes, int n_in,
                              void* d_out, int out_size, void* d_ws, size_t ws_size,
                              hipStream_t stream) {
    const float* feat = (const float*)d_in[0];
    const int*   sidx = (const int*)d_in[1];
    // d_in[2] = num_segments (device scalar, == N_SEG)
    const float* W1 = (const float*)d_in[3];
    const float* b1 = (const float*)d_in[4];
    const float* W2 = (const float*)d_in[5];
    const float* b2 = (const float*)d_in[6];
    const float* W3 = (const float*)d_in[7];
    const float* b3 = (const float*)d_in[8];
    const float* W4 = (const float*)d_in[9];
    const float* b4 = (const float*)d_in[10];

    char* ws = (char*)d_ws;                 // needs ~5.0 MiB
    unsigned* gk   = (unsigned*)ws;         // keys during pass A, decoded f32 before pass B
    __bf16*   W2t  = (__bf16*)(ws + W2T_OFF);
    __bf16*   W3t  = (__bf16*)(ws + W3T_OFF);
    __bf16*   W4t  = (__bf16*)(ws + W4T_OFF);
    unsigned* outk = (unsigned*)d_out;      // keys, decoded in place at the end

    const int n = in_sizes[0] / C_IN;       // 400000

    k_init<<<(N_SEG * C_OUT + 255) / 256, 256, 0, stream>>>(gk, outk);
    k_prep<<<(C_Z * C_CAT + 255) / 256, 256, 0, stream>>>(W2, W3, W4, W2t, W3t, W4t);
    k_passA<<<(n + TA - 1) / TA, 256, 0, stream>>>(feat, sidx, n, W1, b1, W2t, b2, gk);
    k_decode<<<(N_SEG * C_H2 + 255) / 256, 256, 0, stream>>>(gk, N_SEG * C_H2);
    k_passB<<<(n + TB - 1) / TB, 256, SM_TOTAL, stream>>>(feat, sidx, n, W1, b1, W2t, b2,
                                                          (const float*)gk, W3t, b3, W4t, b4, outk);
    k_decode<<<(N_SEG * C_OUT + 255) / 256, 256, 0, stream>>>(outk, N_SEG * C_OUT);
}
